// UCCAEncoder_40553081208841
// MI455X (gfx1250) — matrix operand order repaired and checked by reading, not compiled, not run on hardware
//
#include <hip/hip_runtime.h>

typedef _Float16 half_t;
typedef __attribute__((ext_vector_type(16))) _Float16 v16h;
typedef __attribute__((ext_vector_type(8)))  _Float16 v8h;
typedef __attribute__((ext_vector_type(8)))  float    v8f;
typedef __attribute__((ext_vector_type(4)))  float    v4f;

#define DCH     256
#define NNODES  16384
#define NEDGES  262144
#define EPB     32          // edges (or nodes) per block in the GEMM kernels
#define EF_STRIDE 520       // (512+8) halves -> 1040B row stride, conflict-free b128
#define H1_STRIDE 264       // (256+8) halves -> 528B row stride

// ---- monotonic float<->u32 encoding for atomic max (empty sentinel = 0) ----
__device__ __forceinline__ unsigned enc_f32(float f) {
    int i = __float_as_int(f);
    return (i >= 0) ? ((unsigned)i | 0x80000000u) : ~(unsigned)i;
}
__device__ __forceinline__ float dec_f32(unsigned u) {
    int i = (u & 0x80000000u) ? (int)(u & 0x7FFFFFFFu) : ~(int)u;
    return __int_as_float(i);
}

// load one 16x32 f16 A-fragment worth of data for this lane (2x16B loads)
__device__ __forceinline__ v16h ld_frag16(const half_t* p) {
    v8h lo = *(const v8h*)p;
    v8h hi = *(const v8h*)(p + 16);
    return __builtin_shufflevector(lo, hi, 0,1,2,3,4,5,6,7,8,9,10,11,12,13,14,15);
}

// ---- pack fp32 weight [K][256] into f16 B-fragments ----
// B layout (v_wmma_*_16x16x32_f16, 32x16 B): lane&15 = N, h = lane>>4; element j of v16h
// -> K = 8h + j for j < 8, K = 16 + 8h + (j - 8) for j >= 8.
__global__ void pack_weight_kernel(const float* __restrict__ w,
                                   half_t* __restrict__ dst, int K) {
    int t = blockIdx.x * 256 + threadIdx.x;
    if (t >= K * 256) return;
    int j    = t & 15;
    int lane = (t >> 4) & 31;
    int nt   = (t >> 9) & 15;
    int kc   = t >> 13;
    int k = kc * 32 + ((lane >> 4) << 3) + (j & 7) + ((j >> 3) << 4);
    int n = nt * 16 + (lane & 15);
    dst[t] = (half_t)w[k * 256 + n];
}

// ---- layernorm: fp32 in, f16 out; one wave per node ----
__global__ __launch_bounds__(256) void ln_kernel(const float* __restrict__ x,
                                                 const float* __restrict__ g,
                                                 const float* __restrict__ b,
                                                 half_t* __restrict__ out) {
    int wid  = threadIdx.x >> 5;
    int lane = threadIdx.x & 31;
    int node = blockIdx.x * 8 + wid;
    const float* row = x + (size_t)node * DCH;
    v4f a0 = *(const v4f*)(row + lane * 8);
    v4f a1 = *(const v4f*)(row + lane * 8 + 4);
    float v[8] = {a0[0], a0[1], a0[2], a0[3], a1[0], a1[1], a1[2], a1[3]};
    float s = 0.f;
    for (int i = 0; i < 8; i++) s += v[i];
    for (int m = 16; m >= 1; m >>= 1) s += __shfl_xor(s, m, 32);
    float mu = s * (1.0f / DCH);
    float vs = 0.f;
    for (int i = 0; i < 8; i++) { float d = v[i] - mu; vs += d * d; }
    for (int m = 16; m >= 1; m >>= 1) vs += __shfl_xor(vs, m, 32);
    float rs = rsqrtf(vs * (1.0f / DCH) + 1e-5f);
    v4f g0 = *(const v4f*)(g + lane * 8);
    v4f g1 = *(const v4f*)(g + lane * 8 + 4);
    v4f b0 = *(const v4f*)(b + lane * 8);
    v4f b1 = *(const v4f*)(b + lane * 8 + 4);
    float gg[8] = {g0[0], g0[1], g0[2], g0[3], g1[0], g1[1], g1[2], g1[3]};
    float bb[8] = {b0[0], b0[1], b0[2], b0[3], b1[0], b1[1], b1[2], b1[3]};
    v8h o;
    for (int i = 0; i < 8; i++) o[i] = (half_t)((v[i] - mu) * rs * gg[i] + bb[i]);
    *(v8h*)(out + (size_t)node * DCH + lane * 8) = o;
}

__global__ void fill_u32_kernel(unsigned* __restrict__ p, unsigned v, int n) {
    int i = blockIdx.x * 256 + threadIdx.x;
    if (i < n) p[i] = v;
}

// ---- edge MLP + scatter max: the hot kernel ----
// 128 threads = 4 waves; 32 edges per block; wave -> (mt = wid&1, 8 N-tiles).
__global__ __launch_bounds__(128) void edge_mlp_kernel(
    const half_t* __restrict__ hln,
    const int* __restrict__ srcIdx, const int* __restrict__ dstIdx,
    const half_t* __restrict__ w1p, const float* __restrict__ b1,
    const half_t* __restrict__ w2p, const float* __restrict__ b2,
    unsigned* __restrict__ agg) {
    __shared__ __align__(16) half_t ef[EPB * EF_STRIDE];
    __shared__ __align__(16) half_t h1[EPB * H1_STRIDE];
    __shared__ int ssrc[EPB];
    __shared__ int sdst[EPB];

    int tid = threadIdx.x;
    int ebase = blockIdx.x * EPB;
    if (tid < EPB) {
        ssrc[tid] = srcIdx[ebase + tid];
        sdst[tid] = dstIdx[ebase + tid];
    }
    __syncthreads();

    // build edge features: ef = [x_i, x_j - x_i], x_i = x[dst], x_j = x[src]
    for (int i = tid; i < EPB * DCH; i += 128) {
        int e = i >> 8;
        int c = i & (DCH - 1);
        float xi = (float)hln[(size_t)sdst[e] * DCH + c];
        float xj = (float)hln[(size_t)ssrc[e] * DCH + c];
        ef[e * EF_STRIDE + c]       = (half_t)xi;
        ef[e * EF_STRIDE + DCH + c] = (half_t)(xj - xi);
    }
    __syncthreads();

    int wid  = tid >> 5;
    int lane = tid & 31;
    int mt     = wid & 1;
    int ntBase = (wid >> 1) * 8;
    int lrow = (mt << 4) + (lane & 15);
    int hsel = (lane >> 4) << 3;           // A-frag: K offset 0 or 8
    int rbase = (mt << 4) + ((lane >> 4) << 3);

    // GEMM1: [32,512] x [512,256], K in 16 chunks of 32
    v8f acc[8] = {};
    for (int kc = 0; kc < 16; kc++) {
        v16h a = ld_frag16(&ef[lrow * EF_STRIDE + kc * 32 + hsel]);
        for (int t = 0; t < 8; t++) {
            int nt = ntBase + t;
            v16h bf = *(const v16h*)(w1p + ((((kc * 16 + nt) * 32) + lane) << 4));
            acc[t] = __builtin_amdgcn_wmma_f32_16x16x32_f16(
                false, a, false, bf, (short)0, acc[t], false, false);
        }
    }
    // bias + relu -> h1 (f16, LDS)
    for (int t = 0; t < 8; t++) {
        int ch = (ntBase + t) * 16 + (lane & 15);
        float bb = b1[ch];
        for (int r = 0; r < 8; r++) {
            float v = fmaxf(acc[t][r] + bb, 0.0f);
            h1[(rbase + r) * H1_STRIDE + ch] = (half_t)v;
        }
    }
    __syncthreads();

    // GEMM2: [32,256] x [256,256], K in 8 chunks of 32
    v8f acc2[8] = {};
    for (int kc = 0; kc < 8; kc++) {
        v16h a = ld_frag16(&h1[lrow * H1_STRIDE + kc * 32 + hsel]);
        for (int t = 0; t < 8; t++) {
            int nt = ntBase + t;
            v16h bf = *(const v16h*)(w2p + ((((kc * 16 + nt) * 32) + lane) << 4));
            acc2[t] = __builtin_amdgcn_wmma_f32_16x16x32_f16(
                false, a, false, bf, (short)0, acc2[t], false, false);
        }
    }
    // bias + scatter atomic-max to agg[dst]
    for (int t = 0; t < 8; t++) {
        int ch = (ntBase + t) * 16 + (lane & 15);
        float bb = b2[ch];
        for (int r = 0; r < 8; r++) {
            float v = acc2[t][r] + bb;
            int node = sdst[rbase + r];
            atomicMax(&agg[(size_t)node * DCH + ch], enc_f32(v));
        }
    }
}

// ---- decode agg, relu, add residual ----
__global__ void finalize_kernel(const unsigned* __restrict__ agg,
                                const float* __restrict__ xin,
                                float* __restrict__ xout, int n) {
    int i = blockIdx.x * 256 + threadIdx.x;
    if (i >= n) return;
    unsigned u = agg[i];
    float v = 0.0f;
    if (u != 0u) v = fmaxf(dec_f32(u), 0.0f);
    xout[i] = v + xin[i];
}

// ---- FFN: x += relu(hln @ w1 + b1) @ w2 + b2 ; same WMMA structure ----
__global__ __launch_bounds__(128) void ffn_kernel(
    const half_t* __restrict__ hln,
    const half_t* __restrict__ w1p, const float* __restrict__ b1,
    const half_t* __restrict__ w2p, const float* __restrict__ b2,
    float* __restrict__ x) {
    __shared__ __align__(16) half_t h1[EPB * H1_STRIDE];
    int tid = threadIdx.x;
    int nbase = blockIdx.x * EPB;
    int wid  = tid >> 5;
    int lane = tid & 31;
    int mt     = wid & 1;
    int ntBase = (wid >> 1) * 8;
    int grow = nbase + (mt << 4) + (lane & 15);
    int hsel = (lane >> 4) << 3;
    int rbase = (mt << 4) + ((lane >> 4) << 3);

    v8f acc[8] = {};
    for (int kc = 0; kc < 8; kc++) {
        v16h a = ld_frag16(hln + (size_t)grow * DCH + kc * 32 + hsel);
        for (int t = 0; t < 8; t++) {
            int nt = ntBase + t;
            v16h bf = *(const v16h*)(w1p + ((((kc * 16 + nt) * 32) + lane) << 4));
            acc[t] = __builtin_amdgcn_wmma_f32_16x16x32_f16(
                false, a, false, bf, (short)0, acc[t], false, false);
        }
    }
    for (int t = 0; t < 8; t++) {
        int ch = (ntBase + t) * 16 + (lane & 15);
        float bb = b1[ch];
        for (int r = 0; r < 8; r++) {
            float v = fmaxf(acc[t][r] + bb, 0.0f);
            h1[(rbase + r) * H1_STRIDE + ch] = (half_t)v;
        }
    }
    __syncthreads();

    v8f acc2[8] = {};
    for (int kc = 0; kc < 8; kc++) {
        v16h a = ld_frag16(&h1[((mt << 4) + (lane & 15)) * H1_STRIDE + kc * 32 + hsel]);
        for (int t = 0; t < 8; t++) {
            int nt = ntBase + t;
            v16h bf = *(const v16h*)(w2p + ((((kc * 16 + nt) * 32) + lane) << 4));
            acc2[t] = __builtin_amdgcn_wmma_f32_16x16x32_f16(
                false, a, false, bf, (short)0, acc2[t], false, false);
        }
    }
    for (int t = 0; t < 8; t++) {
        int ch = (ntBase + t) * 16 + (lane & 15);
        float bb = b2[ch];
        for (int r = 0; r < 8; r++) {
            size_t idx = (size_t)(nbase + rbase + r) * DCH + ch;
            x[idx] = acc2[t][r] + bb + x[idx];
        }
    }
}

// ---- gather selected nodes ----
__global__ void gather_kernel(const float* __restrict__ x,
                              const int* __restrict__ sel,
                              float* __restrict__ out) {
    int i = blockIdx.x * 256 + threadIdx.x;     // 8*512*256 total
    int c = i & 255;
    int s = (i >> 8) & 511;
    int b = i >> 17;
    int node = b * 2048 + sel[b * 512 + s];
    out[i] = x[(size_t)node * DCH + c];
}

extern "C" void kernel_launch(void* const* d_in, const int* in_sizes, int n_in,
                              void* d_out, int out_size, void* d_ws, size_t ws_size,
                              hipStream_t stream) {
    const float* x_in      = (const float*)d_in[0];
    const int*   edge      = (const int*)d_in[1];
    const int*   sel       = (const int*)d_in[2];
    const float* conv_w1   = (const float*)d_in[3];
    const float* conv_b1   = (const float*)d_in[4];
    const float* conv_w2   = (const float*)d_in[5];
    const float* conv_b2   = (const float*)d_in[6];
    const float* ln_conv_g = (const float*)d_in[7];
    const float* ln_conv_b = (const float*)d_in[8];
    const float* ln_ffn_g  = (const float*)d_in[9];
    const float* ln_ffn_b  = (const float*)d_in[10];
    const float* ffn_w1    = (const float*)d_in[11];
    const float* ffn_b1    = (const float*)d_in[12];
    const float* ffn_w2    = (const float*)d_in[13];
    const float* ffn_b2    = (const float*)d_in[14];
    float* out = (float*)d_out;

    char* ws = (char*)d_ws;
    float*    x_ws = (float*)ws;    ws += (size_t)NNODES * DCH * 4;
    half_t*   hln  = (half_t*)ws;   ws += (size_t)NNODES * DCH * 2;
    unsigned* agg  = (unsigned*)ws; ws += (size_t)NNODES * DCH * 4;
    half_t*   w1p  = (half_t*)ws;   ws += (size_t)3 * 512 * 256 * 2;
    half_t*   w2p  = (half_t*)ws;   ws += (size_t)3 * 256 * 256 * 2;
    half_t*   fw1p = (half_t*)ws;   ws += (size_t)256 * 256 * 2;
    half_t*   fw2p = (half_t*)ws;   ws += (size_t)256 * 256 * 2;

    for (int l = 0; l < 3; l++) {
        pack_weight_kernel<<<512, 256, 0, stream>>>(conv_w1 + (size_t)l * 512 * 256,
                                                    w1p + (size_t)l * 512 * 256, 512);
        pack_weight_kernel<<<256, 256, 0, stream>>>(conv_w2 + (size_t)l * 256 * 256,
                                                    w2p + (size_t)l * 256 * 256, 256);
    }
    pack_weight_kernel<<<256, 256, 0, stream>>>(ffn_w1, fw1p, 256);
    pack_weight_kernel<<<256, 256, 0, stream>>>(ffn_w2, fw2p, 256);

    const int* srcI = edge;
    const int* dstI = edge + NEDGES;

    const float* xcur = x_in;
    for (int l = 0; l < 3; l++) {
        ln_kernel<<<NNODES / 8, 256, 0, stream>>>(xcur, ln_conv_g, ln_conv_b, hln);
        fill_u32_kernel<<<(NNODES * DCH) / 256, 256, 0, stream>>>(agg, 0u, NNODES * DCH);
        edge_mlp_kernel<<<NEDGES / EPB, 128, 0, stream>>>(
            hln, srcI, dstI,
            w1p + (size_t)l * 512 * 256, conv_b1 + l * DCH,
            w2p + (size_t)l * 256 * 256, conv_b2 + l * DCH, agg);
        finalize_kernel<<<(NNODES * DCH) / 256, 256, 0, stream>>>(agg, xcur, x_ws,
                                                                  NNODES * DCH);
        xcur = x_ws;
    }
    ln_kernel<<<NNODES / 8, 256, 0, stream>>>(x_ws, ln_ffn_g, ln_ffn_b, hln);
    ffn_kernel<<<NNODES / EPB, 128, 0, stream>>>(hln, fw1p, ffn_b1, fw2p, ffn_b2, x_ws);
    gather_kernel<<<(8 * 512 * 256) / 256, 256, 0, stream>>>(x_ws, sel, out);
}
